// FullAttentionLayer_14061722927947
// MI455X (gfx1250) — compile-verified, matrix operand order checked by reading, not run on hardware
//
#include <hip/hip_runtime.h>

// ---------------- constants ----------------
constexpr int SEQ    = 2048;
constexpr int HMODEL = 2048;
constexpr int NHEADS = 16;
constexpr int NKV    = 2;
constexpr int HD     = 256;
constexpr int RDIM   = 64;
constexpr int GROUPS = NHEADS / NKV;   // 8
constexpr int SPAD   = SEQ + 32;       // padded key dim for transposed V
#define ATT_SCALE 0.0625f              // 256^-0.5

typedef __bf16 v16bf __attribute__((ext_vector_type(16)));
typedef float  v8f   __attribute__((ext_vector_type(8)));

union FragB16 { v16bf v; __bf16 e[16]; uint4 q[2]; };

__device__ __forceinline__ __bf16 f2bf(float x) { return (__bf16)x; }

// -------- gfx1250 async global->LDS copy (ASYNCcnt path), feature-gated ----
#if defined(__has_builtin)
#if __has_builtin(__builtin_amdgcn_global_load_async_to_lds_b128)
#define HAVE_ASYNC_LDS 1
#endif
#endif

#ifdef HAVE_ASYNC_LDS
typedef int i4v __attribute__((vector_size(16)));   // matches builtin param pointee
typedef __attribute__((address_space(1))) i4v* as1_i4v_ptr;  // prints as "__device__"
typedef __attribute__((address_space(3))) i4v* as3_i4v_ptr;  // prints as "__shared__"

__device__ __forceinline__ void async_copy_b128(const __bf16* gsrc, __bf16* ldst)
{
    __builtin_amdgcn_global_load_async_to_lds_b128(
        (as1_i4v_ptr)gsrc, (as3_i4v_ptr)ldst, 0, 0);
}
__device__ __forceinline__ void wait_async0()
{
#if __has_builtin(__builtin_amdgcn_s_wait_asynccnt)
    __builtin_amdgcn_s_wait_asynccnt(0);
#else
    asm volatile("s_wait_asynccnt 0x0" ::: "memory");
#endif
}
#endif

// =====================================================================
// Kernel 1: C[M,N] = A[M,K] (f32) * B[N,K]^T (f32), bf16 WMMA math.
// Block = 256 threads (8 waves), 64x64 tile, K-step 32.
// =====================================================================
__global__ __launch_bounds__(256)
void wmma_gemm_f32bf16(const float* __restrict__ A, int lda,
                       const float* __restrict__ B, int ldb,
                       float* __restrict__ C, int ldc,
                       int M, int N, int K)
{
    __shared__ __align__(16) __bf16 As[64 * 32];
    __shared__ __align__(16) __bf16 Bs[64 * 32];

    const int tid  = threadIdx.x;
    const int lane = tid & 31;
    const int wid  = tid >> 5;     // 0..7
    const int wm   = wid >> 1;     // 0..3 -> 16-row strip
    const int wn   = wid & 1;      // 0..1 -> 32-col strip
    const int hl   = lane >> 4;    // half-wave (K-group select per ISA layout)
    const int l16  = lane & 15;

    const int bM = blockIdx.y * 64;
    const int bN = blockIdx.x * 64;

    const int lrow = tid >> 2;          // 0..63
    const int lcol = (tid & 3) * 8;     // 0,8,16,24

    v8f c0 = {}; v8f c1 = {};

    for (int kb = 0; kb < K; kb += 32) {
        // ---- stage A tile (f32 -> bf16) ----
        {
            const float* src = A + (size_t)(bM + lrow) * lda + kb + lcol;
            float4 x0 = *(const float4*)(src);
            float4 x1 = *(const float4*)(src + 4);
            __bf16* dst = &As[lrow * 32 + lcol];
            dst[0] = f2bf(x0.x); dst[1] = f2bf(x0.y); dst[2] = f2bf(x0.z); dst[3] = f2bf(x0.w);
            dst[4] = f2bf(x1.x); dst[5] = f2bf(x1.y); dst[6] = f2bf(x1.z); dst[7] = f2bf(x1.w);
        }
        // ---- stage B tile (f32 -> bf16) ----
        {
            const float* src = B + (size_t)(bN + lrow) * ldb + kb + lcol;
            float4 x0 = *(const float4*)(src);
            float4 x1 = *(const float4*)(src + 4);
            __bf16* dst = &Bs[lrow * 32 + lcol];
            dst[0] = f2bf(x0.x); dst[1] = f2bf(x0.y); dst[2] = f2bf(x0.z); dst[3] = f2bf(x0.w);
            dst[4] = f2bf(x1.x); dst[5] = f2bf(x1.y); dst[6] = f2bf(x1.z); dst[7] = f2bf(x1.w);
        }
        if (kb + 32 < K) {  // global_prefetch_b8 of next tiles
            __builtin_prefetch(A + (size_t)(bM + lrow) * lda + kb + 32 + lcol, 0, 3);
            __builtin_prefetch(B + (size_t)(bN + lrow) * ldb + kb + 32 + lcol, 0, 3);
        }
        __syncthreads();

        // ---- fragments per ISA 16-bit layout: e<8 -> K=hl*8+e, e>=8 -> K=16+hl*8+(e-8)
        FragB16 a, b0, b1;
        const __bf16* ap = &As[(wm * 16 + l16) * 32 + hl * 8];
        a.q[0] = *(const uint4*)(ap);
        a.q[1] = *(const uint4*)(ap + 16);
        const __bf16* bp0 = &Bs[(wn * 32 + l16) * 32 + hl * 8];
        b0.q[0] = *(const uint4*)(bp0);
        b0.q[1] = *(const uint4*)(bp0 + 16);
        const __bf16* bp1 = &Bs[(wn * 32 + 16 + l16) * 32 + hl * 8];
        b1.q[0] = *(const uint4*)(bp1);
        b1.q[1] = *(const uint4*)(bp1 + 16);

        c0 = __builtin_amdgcn_wmma_f32_16x16x32_bf16(false, a.v, false, b0.v, (short)0, c0, false, false);
        c1 = __builtin_amdgcn_wmma_f32_16x16x32_bf16(false, a.v, false, b1.v, (short)0, c1, false, false);
        __syncthreads();
    }

    // ---- store C: VGPR r -> row M = r + hl*8, lane -> col N = l16 ----
#pragma unroll
    for (int r = 0; r < 8; ++r) {
        int row  = bM + wm * 16 + r + hl * 8;
        int col0 = bN + wn * 32 + l16;
        if (row < M) {
            if (col0 < N)      C[(size_t)row * ldc + col0]      = c0[r];
            if (col0 + 16 < N) C[(size_t)row * ldc + col0 + 16] = c1[r];
        }
    }
}

// =====================================================================
// Kernel 2: RMSNorm + RoPE for Q and K, pack bf16; V -> transposed bf16.
// grid = (SEQ, NHEADS + NKV), block = 256 (one thread per head dim)
// =====================================================================
__global__ __launch_bounds__(256)
void norm_rope_pack(const float* __restrict__ qg,
                    const float* __restrict__ kf,
                    const float* __restrict__ vf,
                    const float* __restrict__ cosp,
                    const float* __restrict__ sinp,
                    const float* __restrict__ qnw,
                    const float* __restrict__ knw,
                    __bf16* __restrict__ qbf,
                    __bf16* __restrict__ kbf,
                    __bf16* __restrict__ vT)
{
    __shared__ float red[256];
    __shared__ float ys[256];
    const int s = blockIdx.x;
    const int h = blockIdx.y;
    const int d = threadIdx.x;

    float x;
    const float* w;
    if (h < NHEADS) { x = qg[(size_t)s * 8192 + h * 512 + d]; w = qnw; }
    else            { int kv = h - NHEADS; x = kf[(size_t)s * 512 + kv * 256 + d]; w = knw; }

    red[d] = x * x;
    __syncthreads();
    for (int off = 128; off > 0; off >>= 1) {
        if (d < off) red[d] += red[d + off];
        __syncthreads();
    }
    float inv = rsqrtf(red[0] / 256.0f + 1e-6f);
    float y = x * inv * (1.0f + w[d]);
    ys[d] = y;
    __syncthreads();

    float out = y;
    if (d < RDIM) {
        float c  = cosp[(size_t)s * RDIM + d];
        float sn = sinp[(size_t)s * RDIM + d];
        float rh = (d < 32) ? -ys[d + 32] : ys[d - 32];
        out = y * c + rh * sn;
    }

    if (h < NHEADS) {
        qbf[((size_t)h * SEQ + s) * 256 + d] = f2bf(out);
    } else {
        int kv = h - NHEADS;
        kbf[((size_t)kv * SEQ + s) * 256 + d] = f2bf(out);
        float vv = vf[(size_t)s * 512 + kv * 256 + d];
        vT[((size_t)kv * 256 + d) * SPAD + s] = f2bf(vv);  // transposed for PV B-fragments
    }
}

// =====================================================================
// Kernel 3: causal flash attention, one 16-query tile per block.
// block = 64 threads (2 waves); wave w owns output dims [w*128, w*128+128).
// K tile (32x256 bf16, 16KB) staged block-wide via GLOBAL_LOAD_ASYNC_TO_LDS
// (removes 2x redundant K reads across the two waves); V fragments direct
// from global (disjoint per wave). P round-trips LDS (C->A layout).
// =====================================================================
__global__ __launch_bounds__(64)
void flash_attn(const __bf16* __restrict__ qbf,
                const __bf16* __restrict__ kbf,
                const __bf16* __restrict__ vT,
                const float* __restrict__ qg,
                float* __restrict__ ag)
{
    __shared__ __align__(16) __bf16 plds[2][16 * 32];
#ifdef HAVE_ASYNC_LDS
    __shared__ __align__(16) __bf16 Ks[32 * 256];   // 16KB staged K tile
#endif

    const int tid  = threadIdx.x;
    const int lane = tid & 31;
    const int w    = tid >> 5;     // wave 0/1
    const int hl   = lane >> 4;
    const int l16  = lane & 15;
    const int it   = blockIdx.x;   // query row tile
    const int h    = blockIdx.y;
    const int kv   = h / GROUPS;
    const int qrow0 = it * 16;
    const int qEnd  = qrow0 + 15;
    const int d0    = w * 128;

    // resident Q fragments: 16 rows x 256, 8 A-frags
    FragB16 qf[8];
    {
        const __bf16* qb = qbf + ((size_t)h * SEQ + qrow0 + l16) * 256 + hl * 8;
#pragma unroll
        for (int st = 0; st < 8; ++st) {
            qf[st].q[0] = *(const uint4*)(qb + st * 32);
            qf[st].q[1] = *(const uint4*)(qb + st * 32 + 16);
        }
    }

    v8f zero = {};
    v8f acc[8];
#pragma unroll
    for (int t = 0; t < 8; ++t) acc[t] = zero;
    float m[8], l[8];
#pragma unroll
    for (int r = 0; r < 8; ++r) { m[r] = -1e30f; l[r] = 0.0f; }

    for (int kb = 0; kb <= qEnd; kb += 32) {
        const bool t1 = (kb + 16) <= qEnd;

#ifdef HAVE_ASYNC_LDS
        // ---- block-wide async stage of K rows kb..kb+31 into LDS ----
        // 1024 x 16B chunks; 64 lanes x 16 iterations.
#pragma unroll
        for (int i = 0; i < 16; ++i) {
            int chunk = tid + 64 * i;          // 0..1023
            int row   = chunk >> 5;            // 0..31 local key
            int eoff  = (chunk & 31) * 8;      // bf16 element offset in row
            if (kb + row < SEQ) {              // masked rows may be stale (unused)
                async_copy_b128(kbf + ((size_t)kv * SEQ + kb + row) * 256 + eoff,
                                &Ks[row * 256 + eoff]);
            }
        }
        wait_async0();
        __syncthreads();
#endif

        v8f sc0 = zero, sc1 = zero;
        // ---- QK^T tile 0 (keys kb..kb+15), always active ----
        {
#ifdef HAVE_ASYNC_LDS
            const __bf16* kp = &Ks[l16 * 256 + hl * 8];
#else
            const __bf16* kp = kbf + ((size_t)kv * SEQ + kb + l16) * 256 + hl * 8;
#endif
#pragma unroll
            for (int st = 0; st < 8; ++st) {
                FragB16 bk;
                bk.q[0] = *(const uint4*)(kp + st * 32);
                bk.q[1] = *(const uint4*)(kp + st * 32 + 16);
                sc0 = __builtin_amdgcn_wmma_f32_16x16x32_bf16(false, qf[st].v, false, bk.v, (short)0, sc0, false, false);
            }
        }
        // ---- QK^T tile 1 (keys kb+16..kb+31), only if it intersects causal ----
        if (t1) {
#ifdef HAVE_ASYNC_LDS
            const __bf16* kp = &Ks[(16 + l16) * 256 + hl * 8];
#else
            const __bf16* kp = kbf + ((size_t)kv * SEQ + kb + 16 + l16) * 256 + hl * 8;
#endif
#pragma unroll
            for (int st = 0; st < 8; ++st) {
                FragB16 bk;
                bk.q[0] = *(const uint4*)(kp + st * 32);
                bk.q[1] = *(const uint4*)(kp + st * 32 + 16);
                sc1 = __builtin_amdgcn_wmma_f32_16x16x32_bf16(false, qf[st].v, false, bk.v, (short)0, sc1, false, false);
            }
        }

        // ---- mask + scale + online softmax (row r lives in 16-lane half hl) ----
        float val0[8], val1[8], mx[8];
        const int key0 = kb + l16;
        const int key1 = kb + 16 + l16;
#pragma unroll
        for (int r = 0; r < 8; ++r) {
            int qrow = qrow0 + r + hl * 8;
            val0[r] = sc0[r] * ATT_SCALE + ((key0 <= qrow) ? 0.0f : -10000.0f);
            val1[r] = t1 ? (sc1[r] * ATT_SCALE + ((key1 <= qrow) ? 0.0f : -10000.0f)) : -10000.0f;
            float v = fmaxf(val0[r], val1[r]);
            v = fmaxf(v, __shfl_xor(v, 1, 16));
            v = fmaxf(v, __shfl_xor(v, 2, 16));
            v = fmaxf(v, __shfl_xor(v, 4, 16));
            v = fmaxf(v, __shfl_xor(v, 8, 16));
            mx[r] = v;
        }
        float p0[8], p1[8];
#pragma unroll
        for (int r = 0; r < 8; ++r) {
            float mnew  = fmaxf(m[r], mx[r]);
            float alpha = __expf(m[r] - mnew);
            m[r] = mnew;
            p0[r] = __expf(val0[r] - mnew);
            p1[r] = __expf(val1[r] - mnew);
            float rs = p0[r] + p1[r];
            rs += __shfl_xor(rs, 1, 16);
            rs += __shfl_xor(rs, 2, 16);
            rs += __shfl_xor(rs, 4, 16);
            rs += __shfl_xor(rs, 8, 16);
            l[r] = l[r] * alpha + rs;
#pragma unroll
            for (int t = 0; t < 8; ++t) acc[t][r] *= alpha;
        }

        // ---- P: C-layout -> LDS [16 rows][32 keys] -> A-fragment ----
#pragma unroll
        for (int r = 0; r < 8; ++r) {
            int prow = r + hl * 8;
            plds[w][prow * 32 + l16]      = f2bf(p0[r]);
            plds[w][prow * 32 + 16 + l16] = f2bf(p1[r]);
        }
        FragB16 pf;  // same-wave LDS RAW: DS ops are in-order per wave
        {
            const __bf16* pb = &plds[w][l16 * 32 + hl * 8];
            pf.q[0] = *(const uint4*)(pb);
            pf.q[1] = *(const uint4*)(pb + 16);
        }

        // ---- PV: out[16 x 128] += P[16 x 32] * V[32 x 16] per dim tile ----
#pragma unroll
        for (int t = 0; t < 8; ++t) {
            const __bf16* vp = vT + ((size_t)kv * 256 + d0 + 16 * t + l16) * SPAD + kb + hl * 8;
            FragB16 fv;
            fv.q[0] = *(const uint4*)(vp);
            fv.q[1] = *(const uint4*)(vp + 16);
            acc[t] = __builtin_amdgcn_wmma_f32_16x16x32_bf16(false, pf.v, false, fv.v, (short)0, acc[t], false, false);
        }

#ifdef HAVE_ASYNC_LDS
        __syncthreads();   // protect Ks against WAR from next chunk's staging
#endif
    }

    // ---- epilogue: normalize, gate with sigmoid, store f32 ----
#pragma unroll
    for (int t = 0; t < 8; ++t) {
        int d = d0 + 16 * t + l16;
#pragma unroll
        for (int r = 0; r < 8; ++r) {
            int s = qrow0 + r + hl * 8;
            float o = acc[t][r] / l[r];
            float g = qg[(size_t)s * 8192 + h * 512 + 256 + d];
            float sg = 1.0f / (1.0f + __expf(-g));
            ag[(size_t)s * 4096 + h * 256 + d] = o * sg;
        }
    }
}

// =====================================================================
// Host launcher
// =====================================================================
extern "C" void kernel_launch(void* const* d_in, const int* in_sizes, int n_in,
                              void* d_out, int out_size, void* d_ws, size_t ws_size,
                              hipStream_t stream)
{
    (void)in_sizes; (void)n_in; (void)out_size; (void)ws_size;

    const float* hidden = (const float*)d_in[0];
    const float* cosp   = (const float*)d_in[1];
    const float* sinp   = (const float*)d_in[2];
    const float* qw     = (const float*)d_in[3];  // [8192, 2048]
    const float* kw     = (const float*)d_in[4];  // [512, 2048]
    const float* vw     = (const float*)d_in[5];  // [512, 2048]
    const float* ow     = (const float*)d_in[6];  // [2048, 4096]
    const float* qnw    = (const float*)d_in[7];
    const float* knw    = (const float*)d_in[8];
    float* out = (float*)d_out;

    char* ws = (char*)d_ws;
    size_t off = 0;
    auto alloc = [&](size_t bytes) -> char* {
        char* p = ws + off;
        off += (bytes + 255) & ~(size_t)255;
        return p;
    };
    float*  qg  = (float*)alloc((size_t)SEQ * 8192 * 4);          // fused Q+gate proj
    float*  kf  = (float*)alloc((size_t)SEQ * 512 * 4);
    float*  vf  = (float*)alloc((size_t)SEQ * 512 * 4);
    __bf16* qbf = (__bf16*)alloc((size_t)NHEADS * SEQ * 256 * 2);
    __bf16* kbf = (__bf16*)alloc((size_t)NKV * SEQ * 256 * 2);
    __bf16* vT  = (__bf16*)alloc((size_t)NKV * 256 * SPAD * 2);
    float*  ag  = (float*)alloc((size_t)SEQ * 4096 * 4);          // gated attention out

    dim3 blk256(256);
    // QKV projections (C = hidden * W^T)
    wmma_gemm_f32bf16<<<dim3(8192 / 64, SEQ / 64), blk256, 0, stream>>>(
        hidden, HMODEL, qw, HMODEL, qg, 8192, SEQ, 8192, HMODEL);
    wmma_gemm_f32bf16<<<dim3(512 / 64, SEQ / 64), blk256, 0, stream>>>(
        hidden, HMODEL, kw, HMODEL, kf, 512, SEQ, 512, HMODEL);
    wmma_gemm_f32bf16<<<dim3(512 / 64, SEQ / 64), blk256, 0, stream>>>(
        hidden, HMODEL, vw, HMODEL, vf, 512, SEQ, 512, HMODEL);

    // RMSNorm + RoPE + pack (Q/K), V transpose to bf16
    norm_rope_pack<<<dim3(SEQ, NHEADS + NKV), blk256, 0, stream>>>(
        qg, kf, vf, cosp, sinp, qnw, knw, qbf, kbf, vT);

    // Causal flash attention + sigmoid gating
    flash_attn<<<dim3(SEQ / 16, NHEADS), dim3(64), 0, stream>>>(qbf, kbf, vT, qg, ag);

    // Output projection
    wmma_gemm_f32bf16<<<dim3(HMODEL / 64, SEQ / 64), blk256, 0, stream>>>(
        ag, 4096, ow, 4096, out, HMODEL, SEQ, HMODEL, 4096);
}
